// TransducerJointNetworkV1_26577257628395
// MI455X (gfx1250) — compile-verified, matrix operand order checked by reading, not run on hardware
//
#include <hip/hip_runtime.h>
#include <hip/hip_bf16.h>

typedef __attribute__((ext_vector_type(16))) _Float16 v16h;
typedef __attribute__((ext_vector_type(8)))  _Float16 v8h;
typedef __attribute__((ext_vector_type(8)))  float    v8f;

#define B_ 4
#define T_ 256
#define S_ 128
#define F_ 256
#define H_ 512
#define V_ 1024

#define MTILE   64
#define NBLK    2048           // (B_*T_*S_) / MTILE
#define THREADS 512            // 16 wave32 waves

// ---------------------------------------------------------------------------
// Prep: Asrc[1024][512] = src @ W1[:F] + b1 ;  Ctgt[512][512] = tgt @ W1[F:]
// Written contiguously into ws (Asrc then Ctgt). fp32 accumulate.
// ---------------------------------------------------------------------------
__global__ __launch_bounds__(256) void tjn_prep_proj(
    const float* __restrict__ src, const float* __restrict__ tgt,
    const float* __restrict__ W1, const float* __restrict__ b1,
    float* __restrict__ ws)
{
  int g   = blockIdx.x * 256 + threadIdx.x;   // < 1536*512
  int row = g >> 9;
  int hc  = g & (H_ - 1);
  const float* in;
  const float* w;
  float acc;
  if (row < B_ * T_) {
    in = src + (size_t)row * F_;
    w  = W1 + hc;
    acc = b1[hc];
  } else {
    in = tgt + (size_t)(row - B_ * T_) * F_;
    w  = W1 + (size_t)F_ * H_ + hc;
    acc = 0.0f;
  }
#pragma unroll 8
  for (int k = 0; k < F_; ++k)
    acc = fmaf(in[k], w[(size_t)k * H_], acc);
  ws[g] = acc;
}

// ---------------------------------------------------------------------------
// Pack W2 [512][1024] f32 -> f16 B-fragment layout:
//   W2p[((j*64 + ntile)*32 + lane)*16 + idx]
// per ISA 16-bit B 32x16 layout: lane(0-15)=col, K 0..7 / 16..23 in halves
// 0..7 / 8..15; lane(16-31)=col, K 8..15 / 24..31.
// ---------------------------------------------------------------------------
__global__ __launch_bounds__(256) void tjn_pack_w2(
    const float* __restrict__ W2, _Float16* __restrict__ W2p)
{
  int g   = blockIdx.x * 256 + threadIdx.x;   // < 512*1024
  int k   = g >> 10;
  int col = g & (V_ - 1);
  int j   = k >> 5;
  int ko  = k & 31;
  int nt  = col >> 4;
  int lo  = col & 15;
  int hi  = (ko >> 3) & 1;                    // lane-hi group
  int idx = (ko & 7) | (((ko >> 4) & 1) << 3);
  int lane = lo | (hi << 4);
  size_t p = ((((size_t)j * 64 + nt) * 32 + lane) << 4) + idx;
  W2p[p] = (_Float16)W2[g];
}

// ---------------------------------------------------------------------------
// Main: per block, 64 rows x full V=1024.
//   h = relu(Asrc[bt] + Ctgt[b*S+s])  (built once in LDS as f16, swizzled)
//   logits = h @ W2p + b2             (v_wmma_f32_16x16x32_f16)
//   out = log_softmax(logits)         (wave butterfly + LDS reductions,
//                                      non-temporal streaming stores)
// ---------------------------------------------------------------------------
__global__ __launch_bounds__(THREADS) void tjn_joint(
    const float* __restrict__ Asrc, const float* __restrict__ Ctgt,
    const _Float16* __restrict__ W2p, const float* __restrict__ b2,
    float* __restrict__ out)
{
  __shared__ __align__(16) unsigned char smem[65536];
  v8h*   hb8  = (v8h*)smem;                 // 64 rows x 64 v8h chunks (swizzled)
  float* sred = (float*)smem;               // aliased after GEMM: [16][64]
  float* gred = (float*)(smem + 4096);      // [64]

  const int tid  = threadIdx.x;
  const int lane = tid & 31;
  const int wave = tid >> 5;
  const int sw   = lane & 15;
  const int hi   = lane >> 4;               // 0 or 1
  const int rb   = hi * 8;

  const int R  = blockIdx.x * MTILE;        // global row base
  const int bt = R >> 7;                    // / S_
  const int b  = bt >> 8;                   // / T_
  const int s0 = R & (S_ - 1);

  // ---- build 64x512 f16 h-tile in LDS (XOR chunk swizzle, bank-conflict free)
  for (int t = tid; t < MTILE * 16; t += THREADS) {
    int row = t & (MTILE - 1);
    int c   = t >> 6;                       // 32-half chunk, 0..15
    int ctr = b * S_ + s0 + row;
    const float4* a4 = (const float4*)(Asrc + (size_t)bt  * H_ + c * 32);
    const float4* c4 = (const float4*)(Ctgt + (size_t)ctr * H_ + c * 32);
    v8h hv[4];
#pragma unroll
    for (int i = 0; i < 8; ++i) {
      float4 av = a4[i], tv = c4[i];
      int e = i * 4;
      hv[e >> 3][(e & 7) + 0] = (_Float16)fmaxf(av.x + tv.x, 0.0f);
      hv[e >> 3][(e & 7) + 1] = (_Float16)fmaxf(av.y + tv.y, 0.0f);
      hv[e >> 3][(e & 7) + 2] = (_Float16)fmaxf(av.z + tv.z, 0.0f);
      hv[e >> 3][(e & 7) + 3] = (_Float16)fmaxf(av.w + tv.w, 0.0f);
    }
    int rsw = row & 15;
#pragma unroll
    for (int q = 0; q < 4; ++q)
      hb8[row * 64 + ((c * 4 + q) ^ rsw)] = hv[q];
  }
  __syncthreads();

  // ---- GEMM: wave covers 4 m-tiles x 4 n-tiles, K=512 in 16 steps of 32
  v8f acc[4][4];
#pragma unroll
  for (int m = 0; m < 4; ++m)
#pragma unroll
    for (int i = 0; i < 4; ++i)
      acc[m][i] = {};

  const v8h* w2v = (const v8h*)W2p;

#pragma unroll 4
  for (int j = 0; j < 16; ++j) {
    v16h af[4];
#pragma unroll
    for (int m = 0; m < 4; ++m) {
      int base = (m * 16 + sw) * 64;
      v8h alo = hb8[base + ((4 * j + hi)     ^ sw)];
      v8h ahi = hb8[base + ((4 * j + 2 + hi) ^ sw)];
      af[m] = __builtin_shufflevector(alo, ahi,
              0,1,2,3,4,5,6,7,8,9,10,11,12,13,14,15);
    }
#pragma unroll
    for (int i = 0; i < 4; ++i) {
      int nt = wave * 4 + i;
      size_t bo = (((size_t)j * 64 + nt) * 32 + lane) * 2;
      v8h blo = w2v[bo];
      v8h bhi = w2v[bo + 1];
      v16h bf = __builtin_shufflevector(blo, bhi,
                0,1,2,3,4,5,6,7,8,9,10,11,12,13,14,15);
#pragma unroll
      for (int m = 0; m < 4; ++m)
        acc[m][i] = __builtin_amdgcn_wmma_f32_16x16x32_f16(
            false, af[m], false, bf, (short)0, acc[m][i], false, false);
    }
  }
  __syncthreads();          // h-tile dead; smem now reduction scratch

  // ---- add b2 (lane's column is fixed per n-tile)
  float bias[4];
#pragma unroll
  for (int i = 0; i < 4; ++i) bias[i] = b2[wave * 64 + i * 16 + sw];
#pragma unroll
  for (int m = 0; m < 4; ++m)
#pragma unroll
    for (int i = 0; i < 4; ++i)
#pragma unroll
      for (int r = 0; r < 8; ++r)
        acc[m][i][r] += bias[i];

  // ---- pass 1: row max (lane-local -> 16-lane butterfly -> cross-wave LDS)
  float mx[4][8];
#pragma unroll
  for (int m = 0; m < 4; ++m)
#pragma unroll
    for (int r = 0; r < 8; ++r)
      mx[m][r] = fmaxf(fmaxf(acc[m][0][r], acc[m][1][r]),
                       fmaxf(acc[m][2][r], acc[m][3][r]));
  for (int off = 1; off < 16; off <<= 1)
#pragma unroll
    for (int m = 0; m < 4; ++m)
#pragma unroll
      for (int r = 0; r < 8; ++r)
        mx[m][r] = fmaxf(mx[m][r], __shfl_xor(mx[m][r], off, 32));
  if (sw == 0) {
#pragma unroll
    for (int m = 0; m < 4; ++m)
#pragma unroll
      for (int r = 0; r < 8; ++r)
        sred[wave * 64 + m * 16 + rb + r] = mx[m][r];
  }
  __syncthreads();
  if (tid < 64) {
    float g = sred[tid];
#pragma unroll
    for (int w = 1; w < 16; ++w) g = fmaxf(g, sred[w * 64 + tid]);
    gred[tid] = g;
  }
  __syncthreads();
#pragma unroll
  for (int m = 0; m < 4; ++m)
#pragma unroll
    for (int r = 0; r < 8; ++r)
      mx[m][r] = gred[m * 16 + rb + r];

  // ---- pass 2: sum of exp(x - max)
  float sm[4][8];
#pragma unroll
  for (int m = 0; m < 4; ++m)
#pragma unroll
    for (int r = 0; r < 8; ++r) {
      float s = 0.0f;
#pragma unroll
      for (int i = 0; i < 4; ++i) s += __expf(acc[m][i][r] - mx[m][r]);
      sm[m][r] = s;
    }
  for (int off = 1; off < 16; off <<= 1)
#pragma unroll
    for (int m = 0; m < 4; ++m)
#pragma unroll
      for (int r = 0; r < 8; ++r)
        sm[m][r] += __shfl_xor(sm[m][r], off, 32);
  if (sw == 0) {
#pragma unroll
    for (int m = 0; m < 4; ++m)
#pragma unroll
      for (int r = 0; r < 8; ++r)
        sred[wave * 64 + m * 16 + rb + r] = sm[m][r];
  }
  __syncthreads();
  if (tid < 64) {
    float s = 0.0f;
#pragma unroll
    for (int w = 0; w < 16; ++w) s += sred[w * 64 + tid];
    gred[tid] = __logf(s);
  }
  __syncthreads();

  // ---- out = logits - max - log(sumexp); stream with non-temporal stores so
  // the 512MB write-once output doesn't evict W2p/Asrc/Ctgt from L2.
  float offv[4][8];
#pragma unroll
  for (int m = 0; m < 4; ++m)
#pragma unroll
    for (int r = 0; r < 8; ++r)
      offv[m][r] = mx[m][r] + gred[m * 16 + rb + r];

  float* obase = out + (size_t)R * V_ + wave * 64 + sw;
#pragma unroll
  for (int m = 0; m < 4; ++m)
#pragma unroll
    for (int i = 0; i < 4; ++i)
#pragma unroll
      for (int r = 0; r < 8; ++r)
        __builtin_nontemporal_store(
            acc[m][i][r] - offv[m][r],
            obase + (size_t)(m * 16 + rb + r) * V_ + i * 16);
}

// ---------------------------------------------------------------------------
extern "C" void kernel_launch(void* const* d_in, const int* in_sizes, int n_in,
                              void* d_out, int out_size, void* d_ws, size_t ws_size,
                              hipStream_t stream) {
  const float* src = (const float*)d_in[0];
  const float* tgt = (const float*)d_in[1];
  const float* W1  = (const float*)d_in[2];
  const float* b1  = (const float*)d_in[3];
  const float* W2  = (const float*)d_in[4];
  const float* b2  = (const float*)d_in[5];
  float* out = (float*)d_out;

  float* ws = (float*)d_ws;
  float* Asrc = ws;                                   // 1024*512 f32
  float* Ctgt = ws + (size_t)B_ * T_ * H_;            // 512*512 f32
  _Float16* W2p = (_Float16*)((char*)d_ws +
      ((size_t)B_ * T_ * H_ + (size_t)B_ * S_ * H_) * sizeof(float)); // 1 MB f16

  // projections: (B*T + B*S) * H outputs = 1536*512 -> 3072 blocks of 256
  tjn_prep_proj<<<3072, 256, 0, stream>>>(src, tgt, W1, b1, ws);
  // pack W2: 512*1024 elems -> 2048 blocks of 256
  tjn_pack_w2<<<2048, 256, 0, stream>>>(W2, W2p);
  // main fused GEMM + log_softmax
  tjn_joint<<<NBLK, THREADS, 0, stream>>>(Asrc, Ctgt, W2p, b2, out);
}
